// SketchRNN_87419764343375
// MI455X (gfx1250) — matrix operand order repaired and checked by reading, not compiled, not run on hardware
//
#include <hip/hip_runtime.h>
#include <hip/hip_bf16.h>
#include <math.h>

#define Bn   1024
#define Tn   256
#define Hn   512
#define LATn 128
#define EMBn 64
#define G3H  1536   // 3*H
#define DIN0 195    // decoder layer-0 input width (3 + LAT + EMB)
#define GOFF ((size_t)Hn * Hn)   // element offset between gate blocks of W

typedef __attribute__((ext_vector_type(16))) __bf16 v16bf;
typedef __attribute__((ext_vector_type(8)))  float  v8f;

static __device__ __forceinline__ __bf16 f2bf(float f) {
    union { float f; unsigned u; } v; v.f = f;
    unsigned r = (v.u + 0x7FFFu + ((v.u >> 16) & 1u)) >> 16;   // RNE
    union { unsigned short s; __bf16 b; } o; o.s = (unsigned short)r;
    return o.b;
}

static __device__ __forceinline__ float sigf(float x) { return 1.0f / (1.0f + __expf(-x)); }

// ---------------------------------------------------------------------------
// fp32 -> bf16 weight conversion (row-major preserved)
// ---------------------------------------------------------------------------
__global__ void k_f2bf(const float* __restrict__ src, __bf16* __restrict__ dst, int n) {
    int i = blockIdx.x * blockDim.x + threadIdx.x;
    if (i < n) dst[i] = f2bf(src[i]);
}

// ---------------------------------------------------------------------------
// K-loop over one 512-wide operand.
// A: pre-packed bf16 in WMMA A-operand layout -> one coalesced 32B load/lane.
//    aBase = apk + (tileM*512 + lane)*16 ; per-chunk stride 512 elements.
// B: weight rows (gate columns), 16 contiguous bf16 per lane; the three gate
//    blocks are constant offsets (GOFF elements) off one base pointer.
// ---------------------------------------------------------------------------
static __device__ __forceinline__ void wmma_accum(
    const __bf16* __restrict__ aBase,
    const __bf16* __restrict__ bRow,    // W + ncol*Hn + half*16
    v8f& accR, v8f& accZ, v8f& accN)
{
    for (int kc = 0; kc < Hn / 32; ++kc) {
        v16bf a  = *(const v16bf*)(aBase + (size_t)kc * 512);
        v16bf br = *(const v16bf*)(bRow + kc * 32);
        v16bf bz = *(const v16bf*)(bRow + GOFF + kc * 32);
        v16bf bn = *(const v16bf*)(bRow + 2 * GOFF + kc * 32);
        accR = __builtin_amdgcn_wmma_f32_16x16x32_bf16(false, a, false, br, (short)0, accR, false, false);
        accZ = __builtin_amdgcn_wmma_f32_16x16x32_bf16(false, a, false, bz, (short)0, accZ, false, false);
        accN = __builtin_amdgcn_wmma_f32_16x16x32_bf16(false, a, false, bn, (short)0, accN, false, false);
    }
}

// ---------------------------------------------------------------------------
// Fused GRU step: gates = h@Whh^T [+ x2@Wih^T] [+ cgi] [+ xs(3)@Wsm^T]
//                 [+ b_ih] + b_hh ; gate math + state update + bf16 re-pack.
// Grid: (Bn/16, Hn/64), block 128 (4 waves, one 16-col slice each).
// All branches are wave-uniform (scalar args) -> EXEC all-ones at each WMMA.
// ---------------------------------------------------------------------------
__global__ __launch_bounds__(128) void k_gru_step(
    const __bf16* __restrict__ hpk,     // packed bf16 state (A layout)
    const __bf16* __restrict__ x2pk,    // optional packed input-path source
    const __bf16* __restrict__ Whh,     // [1536,512] bf16 row-major
    const __bf16* __restrict__ Wih,     // [1536,512] bf16 row-major (with x2)
    const float*  __restrict__ cgi,     // optional [B,1536] precomputed i-gates
    const float*  __restrict__ b_ih,    // optional [1536]
    const float*  __restrict__ b_hh,    // [1536]
    const float*  __restrict__ xs,      // optional [B, ld_xs] 3-col input
    int ld_xs,
    const float*  __restrict__ Wsm,     // optional fp32 [1536, ld_wsm], cols 0..2
    int ld_wsm,
    const float*  __restrict__ h_in,    // fp32 state (for z-blend)
    float*        __restrict__ h_out,   // fp32 state out
    __bf16*       __restrict__ h_out_pk)// packed bf16 state out (A layout)
{
    const int lane = threadIdx.x & 31;
    const int wave = threadIdx.x >> 5;
    const int half = lane >> 4;
    const int l16  = lane & 15;
    const int m0   = blockIdx.x * 16;
    const int n0   = blockIdx.y * 64 + wave * 16;
    const int ncol = n0 + l16;

    v8f accR = {}, accZ = {}, accNh = {}, accNi = {};

    const __bf16* aBase = hpk + ((size_t)blockIdx.x * 512 + lane) * 16;

    // hidden path: r, z, n(h) -- n(h) kept separate for r * h_n
    wmma_accum(aBase, Whh + (size_t)ncol * Hn + half * 16, accR, accZ, accNh);

    // input path (layer-1 style, in = H): r, z accumulate; n(i) separate
    if (x2pk) {
        const __bf16* a2 = x2pk + ((size_t)blockIdx.x * 512 + lane) * 16;
        wmma_accum(a2, Wih + (size_t)ncol * Hn + half * 16, accR, accZ, accNi);
    }

    const float bhR = b_hh[ncol], bhZ = b_hh[Hn + ncol], bhN = b_hh[2 * Hn + ncol];
    float biR = 0.f, biZ = 0.f, biN = 0.f;
    if (b_ih) { biR = b_ih[ncol]; biZ = b_ih[Hn + ncol]; biN = b_ih[2 * Hn + ncol]; }

    float wr0 = 0.f, wr1 = 0.f, wr2 = 0.f;
    float wz0 = 0.f, wz1 = 0.f, wz2 = 0.f;
    float wn0 = 0.f, wn1 = 0.f, wn2 = 0.f;
    if (Wsm) {
        const float* p;
        p = Wsm + (size_t)(0 * Hn + ncol) * ld_wsm; wr0 = p[0]; wr1 = p[1]; wr2 = p[2];
        p = Wsm + (size_t)(1 * Hn + ncol) * ld_wsm; wz0 = p[0]; wz1 = p[1]; wz2 = p[2];
        p = Wsm + (size_t)(2 * Hn + ncol) * ld_wsm; wn0 = p[0]; wn1 = p[1]; wn2 = p[2];
    }

    // packed-store address components: this lane's output column = next-step K
    const int kk  = ncol & 31;
    const int lo  = kk & 15;
    const int hp_ = kk >> 4;                        // lane-half in A layout
    const int e_  = lo;                             // element within v16bf
    const size_t pkBase =
        (((size_t)blockIdx.x * 16 + (ncol >> 5)) * 32 + (size_t)hp_ * 16) * 16 + e_;

    #pragma unroll
    for (int v = 0; v < 8; ++v) {
        const int row = m0 + half * 8 + v;        // C/D layout: M = v + 8*(lane/16)
        const int m   = half * 8 + v;             // row within tile
        float gr = accR[v]  + bhR + biR;
        float gz = accZ[v]  + bhZ + biZ;
        float hn = accNh[v] + bhN;                // hidden-path n-gate (gets * r)
        float in = accNi[v] + biN;                // input-path n-gate
        if (cgi) {
            const float* c = cgi + (size_t)row * G3H;
            gr += c[ncol]; gz += c[Hn + ncol]; in += c[2 * Hn + ncol];
        }
        if (Wsm) {
            const float* xp = xs + (size_t)row * ld_xs;
            const float xv0 = xp[0], xv1 = xp[1], xv2 = xp[2];
            gr += xv0 * wr0 + xv1 * wr1 + xv2 * wr2;
            gz += xv0 * wz0 + xv1 * wz1 + xv2 * wz2;
            in += xv0 * wn0 + xv1 * wn1 + xv2 * wn2;
        }
        const float r    = sigf(gr);
        const float zg   = sigf(gz);
        const float nn   = tanhf(in + r * hn);
        const float hprv = h_in[(size_t)row * Hn + ncol];
        const float hnew = (1.f - zg) * nn + zg * hprv;
        h_out[(size_t)row * Hn + ncol] = hnew;
        h_out_pk[pkBase + (size_t)m * 16] = f2bf(hnew);
    }
}

// ---------------------------------------------------------------------------
// Decoder time-invariant input gates: cgi[b,g] = b_ih0[g] + z@Wih0[:,3:131]^T
//                                              + emb[label]@Wih0[:,131:195]^T
// ---------------------------------------------------------------------------
__global__ void k_cgi0(const float* __restrict__ z, const float* __restrict__ emb,
                       const int* __restrict__ labels,
                       const float* __restrict__ Wih0, const float* __restrict__ b_ih0,
                       float* __restrict__ cgi)
{
    int idx = blockIdx.x * blockDim.x + threadIdx.x;
    if (idx >= Bn * G3H) return;
    const int b = idx / G3H, g = idx - b * G3H;
    const float* wr = Wih0 + (size_t)g * DIN0;
    float s = b_ih0[g];
    const float* zp = z + (size_t)b * LATn;
    for (int k = 0; k < LATn; ++k) s += zp[k] * wr[3 + k];
    const float* lp = emb + (size_t)labels[b] * EMBn;
    for (int k = 0; k < EMBn; ++k) s += lp[k] * wr[3 + LATn + k];
    cgi[idx] = s;
}

// ---------------------------------------------------------------------------
// Latent heads: mu, clipped log_var, z
// ---------------------------------------------------------------------------
__global__ void k_latent(const float* __restrict__ hlast,
                         const float* __restrict__ Wmu, const float* __restrict__ bmu,
                         const float* __restrict__ Wlv, const float* __restrict__ blv,
                         const float* __restrict__ eps,
                         float* __restrict__ mu_out, float* __restrict__ lv_out,
                         float* __restrict__ z_out)
{
    int idx = blockIdx.x * blockDim.x + threadIdx.x;
    if (idx >= Bn * LATn) return;
    const int b = idx / LATn, l = idx - b * LATn;
    const float* h  = hlast + (size_t)b * Hn;
    const float* wm = Wmu + (size_t)l * Hn;
    const float* wl = Wlv + (size_t)l * Hn;
    float sm = bmu[l], sl = blv[l];
    for (int k = 0; k < Hn; ++k) { const float hv = h[k]; sm += hv * wm[k]; sl += hv * wl[k]; }
    sl = fminf(2.f, fmaxf(-5.f, sl));
    mu_out[idx] = sm;
    lv_out[idx] = sl;
    z_out[idx]  = sm + eps[idx] * __expf(0.5f * sl);
}

__global__ void k_kld(const float* __restrict__ mu, const float* __restrict__ lv,
                      float* __restrict__ kld)
{
    int b = blockIdx.x * blockDim.x + threadIdx.x;
    if (b >= Bn) return;
    float s = 0.f;
    for (int l = 0; l < LATn; ++l) {
        const float m = mu[(size_t)b * LATn + l];
        const float v = lv[(size_t)b * LATn + l];
        s += 1.f + v - m * m - __expf(v);
    }
    kld[b] = -0.5f * s;
}

// ---------------------------------------------------------------------------
// Decoder output head + feedback + running recon-loss accumulation
// ---------------------------------------------------------------------------
__global__ void k_dec_out(const float* __restrict__ hB,
                          const float* __restrict__ Wo, const float* __restrict__ bo,
                          const float* __restrict__ x, int t,
                          float* __restrict__ prev_out, float* __restrict__ recon,
                          float* __restrict__ loss_acc)
{
    int b = blockIdx.x * blockDim.x + threadIdx.x;
    if (b >= Bn) return;
    const float* h = hB + (size_t)b * Hn;
    float o0 = bo[0], o1 = bo[1], o2 = bo[2];
    for (int k = 0; k < Hn; ++k) {
        const float hv = h[k];
        o0 += hv * Wo[k]; o1 += hv * Wo[Hn + k]; o2 += hv * Wo[2 * Hn + k];
    }
    const size_t ro = ((size_t)b * Tn + t) * 3;
    recon[ro] = o0; recon[ro + 1] = o1; recon[ro + 2] = o2;
    prev_out[b * 3] = o0; prev_out[b * 3 + 1] = o1; prev_out[b * 3 + 2] = o2;
    const float d0 = o0 - x[ro], d1 = o1 - x[ro + 1], d2 = o2 - x[ro + 2];
    loss_acc[b] += d0 * d0 + d1 * d1 + d2 * d2;
}

__global__ void k_fin_loss(const float* __restrict__ loss_acc, float* __restrict__ recon_loss)
{
    int b = blockIdx.x * blockDim.x + threadIdx.x;
    if (b < Bn) recon_loss[b] = loss_acc[b] * (1.0f / (float)(Tn * 3));
}

// ---------------------------------------------------------------------------
extern "C" void kernel_launch(void* const* d_in, const int* in_sizes, int n_in,
                              void* d_out, int out_size, void* d_ws, size_t ws_size,
                              hipStream_t stream)
{
    (void)in_sizes; (void)n_in; (void)out_size; (void)ws_size;

    const float* x      = (const float*)d_in[0];
    const float* eps    = (const float*)d_in[1];
    const int*   labels = (const int*)d_in[2];
    // d_in[3] = max_seq_len (Tn is compile-time)
    const float* eW_ih0 = (const float*)d_in[4];
    const float* eW_hh0 = (const float*)d_in[5];
    const float* eb_ih0 = (const float*)d_in[6];
    const float* eb_hh0 = (const float*)d_in[7];
    const float* eW_ih1 = (const float*)d_in[8];
    const float* eW_hh1 = (const float*)d_in[9];
    const float* eb_ih1 = (const float*)d_in[10];
    const float* eb_hh1 = (const float*)d_in[11];
    const float* dW_ih0 = (const float*)d_in[12];
    const float* dW_hh0 = (const float*)d_in[13];
    const float* db_ih0 = (const float*)d_in[14];
    const float* db_hh0 = (const float*)d_in[15];
    const float* dW_ih1 = (const float*)d_in[16];
    const float* dW_hh1 = (const float*)d_in[17];
    const float* db_ih1 = (const float*)d_in[18];
    const float* db_hh1 = (const float*)d_in[19];
    const float* fmu_W  = (const float*)d_in[20];
    const float* fmu_b  = (const float*)d_in[21];
    const float* flv_W  = (const float*)d_in[22];
    const float* flv_b  = (const float*)d_in[23];
    const float* fo_W   = (const float*)d_in[24];
    const float* fo_b   = (const float*)d_in[25];
    const float* emb    = (const float*)d_in[26];

    float* out     = (float*)d_out;
    float* o_recon = out;
    float* o_mu    = out + (size_t)Bn * Tn * 3;
    float* o_lv    = o_mu + (size_t)Bn * LATn;
    float* o_rl    = o_lv + (size_t)Bn * LATn;
    float* o_kl    = o_rl + Bn;

    // ---- carve workspace ----
    char* p = (char*)d_ws;
    auto carve = [&](size_t bytes) -> char* {
        char* r = p; p += (bytes + 255) & ~(size_t)255; return r;
    };
    const size_t WN = (size_t)G3H * Hn;              // weight elems
    __bf16* wbEhh0 = (__bf16*)carve(WN * 2);
    __bf16* wbEih1 = (__bf16*)carve(WN * 2);
    __bf16* wbEhh1 = (__bf16*)carve(WN * 2);
    __bf16* wbDhh0 = (__bf16*)carve(WN * 2);
    __bf16* wbDih1 = (__bf16*)carve(WN * 2);
    __bf16* wbDhh1 = (__bf16*)carve(WN * 2);
    const size_t HB  = (size_t)Bn * Hn * sizeof(float);
    const size_t HPK = (size_t)Bn * Hn * sizeof(__bf16);
    float* hE0a = (float*)carve(HB); float* hE0b = (float*)carve(HB);
    float* hE1a = (float*)carve(HB); float* hE1b = (float*)carve(HB);
    float* hAa  = (float*)carve(HB); float* hAb  = (float*)carve(HB);
    float* hBa  = (float*)carve(HB); float* hBb  = (float*)carve(HB);
    __bf16* pE0a = (__bf16*)carve(HPK); __bf16* pE0b = (__bf16*)carve(HPK);
    __bf16* pE1a = (__bf16*)carve(HPK); __bf16* pE1b = (__bf16*)carve(HPK);
    __bf16* pAa  = (__bf16*)carve(HPK); __bf16* pAb  = (__bf16*)carve(HPK);
    __bf16* pBa  = (__bf16*)carve(HPK); __bf16* pBb  = (__bf16*)carve(HPK);
    float* zbuf = (float*)carve((size_t)Bn * LATn * sizeof(float));
    float* cgi0 = (float*)carve((size_t)Bn * G3H * sizeof(float));
    float* prev = (float*)carve((size_t)Bn * 3 * sizeof(float));
    float* lacc = (float*)carve((size_t)Bn * sizeof(float));

    // ---- weight conversion fp32 -> bf16 ----
    {
        const int n = (int)WN, blk = 256, g = (n + blk - 1) / blk;
        k_f2bf<<<g, blk, 0, stream>>>(eW_hh0, wbEhh0, n);
        k_f2bf<<<g, blk, 0, stream>>>(eW_ih1, wbEih1, n);
        k_f2bf<<<g, blk, 0, stream>>>(eW_hh1, wbEhh1, n);
        k_f2bf<<<g, blk, 0, stream>>>(dW_hh0, wbDhh0, n);
        k_f2bf<<<g, blk, 0, stream>>>(dW_ih1, wbDih1, n);
        k_f2bf<<<g, blk, 0, stream>>>(dW_hh1, wbDhh1, n);
    }

    // ---- zero-init states & accumulators ----
    hipMemsetAsync(hE0a, 0, HB, stream); hipMemsetAsync(hE0b, 0, HB, stream);
    hipMemsetAsync(hE1a, 0, HB, stream); hipMemsetAsync(hE1b, 0, HB, stream);
    hipMemsetAsync(hAa,  0, HB, stream); hipMemsetAsync(hAb,  0, HB, stream);
    hipMemsetAsync(hBa,  0, HB, stream); hipMemsetAsync(hBb,  0, HB, stream);
    hipMemsetAsync(pE0a, 0, HPK, stream); hipMemsetAsync(pE0b, 0, HPK, stream);
    hipMemsetAsync(pE1a, 0, HPK, stream); hipMemsetAsync(pE1b, 0, HPK, stream);
    hipMemsetAsync(pAa,  0, HPK, stream); hipMemsetAsync(pAb,  0, HPK, stream);
    hipMemsetAsync(pBa,  0, HPK, stream); hipMemsetAsync(pBb,  0, HPK, stream);
    hipMemsetAsync(prev, 0, (size_t)Bn * 3 * sizeof(float), stream);
    hipMemsetAsync(lacc, 0, (size_t)Bn * sizeof(float), stream);

    const dim3 gGru(Bn / 16, Hn / 64);
    const int  bGru = 128;

    // ---- encoder: layer0 (3-col input path) + layer1 (fused i+h GEMMs) ----
    float  *h0c = hE0a, *h0n = hE0b, *h1c = hE1a, *h1n = hE1b;
    __bf16 *p0c = pE0a, *p0n = pE0b, *p1c = pE1a, *p1n = pE1b;
    for (int t = 0; t < Tn; ++t) {
        k_gru_step<<<gGru, bGru, 0, stream>>>(
            p0c, nullptr, wbEhh0, nullptr,
            nullptr, eb_ih0, eb_hh0,
            x + (size_t)t * 3, Tn * 3, eW_ih0, 3,
            h0c, h0n, p0n);
        k_gru_step<<<gGru, bGru, 0, stream>>>(
            p1c, p0n, wbEhh1, wbEih1,
            nullptr, eb_ih1, eb_hh1,
            nullptr, 0, nullptr, 0,
            h1c, h1n, p1n);
        float* tf; __bf16* tb;
        tf = h0c; h0c = h0n; h0n = tf;  tb = p0c; p0c = p0n; p0n = tb;
        tf = h1c; h1c = h1n; h1n = tf;  tb = p1c; p1c = p1n; p1n = tb;
    }
    const float* h_last = h1c;

    // ---- latent: mu / log_var / z / kld ----
    {
        const int n = Bn * LATn, blk = 256;
        k_latent<<<(n + blk - 1) / blk, blk, 0, stream>>>(
            h_last, fmu_W, fmu_b, flv_W, flv_b, eps, o_mu, o_lv, zbuf);
        k_kld<<<(Bn + blk - 1) / blk, blk, 0, stream>>>(o_mu, o_lv, o_kl);
    }

    // ---- decoder constant input-gate tensor ----
    {
        const int n = Bn * G3H, blk = 256;
        k_cgi0<<<(n + blk - 1) / blk, blk, 0, stream>>>(zbuf, emb, labels, dW_ih0, db_ih0, cgi0);
    }

    // ---- decoder loop ----
    float  *hAc = hAa, *hAn = hAb, *hBc = hBa, *hBn = hBb;
    __bf16 *pAc = pAa, *pAn = pAb, *pBc = pBa, *pBn = pBb;
    for (int t = 0; t < Tn; ++t) {
        if (t == 0) {
            // first input is all-zeros: gi = b_ih0 only
            k_gru_step<<<gGru, bGru, 0, stream>>>(
                pAc, nullptr, wbDhh0, nullptr,
                nullptr, db_ih0, db_hh0,
                nullptr, 0, nullptr, 0,
                hAc, hAn, pAn);
        } else {
            // gi = prev_out(3 cols)@W[:, :3] + cgi0 (z/emb/bias folded)
            k_gru_step<<<gGru, bGru, 0, stream>>>(
                pAc, nullptr, wbDhh0, nullptr,
                cgi0, nullptr, db_hh0,
                prev, 3, dW_ih0, DIN0,
                hAc, hAn, pAn);
        }
        k_gru_step<<<gGru, bGru, 0, stream>>>(
            pBc, pAn, wbDhh1, wbDih1,
            nullptr, db_ih1, db_hh1,
            nullptr, 0, nullptr, 0,
            hBc, hBn, pBn);
        k_dec_out<<<(Bn + 255) / 256, 256, 0, stream>>>(
            hBn, fo_W, fo_b, x, t, prev, o_recon, lacc);
        float* tf; __bf16* tb;
        tf = hAc; hAc = hAn; hAn = tf;  tb = pAc; pAc = pAn; pAn = tb;
        tf = hBc; hBc = hBn; hBn = tf;  tb = pBc; pBc = pBn; pBn = tb;
    }

    k_fin_loss<<<(Bn + 255) / 256, 256, 0, stream>>>(lacc, o_rl);
}
